// SlidingWindowAttention_38920993636957
// MI455X (gfx1250) — matrix operand order repaired and checked by reading, not compiled, not run on hardware
//
#include <hip/hip_runtime.h>

// ---------------------------------------------------------------------------
// Sliding-window attention for MI455X (gfx1250, wave32, WMMA).
//   kernel 1: qkv = x @ w_qkv          (f32 in, f16 WMMA, f16 out -> ws)
//   kernel 2: flash-style windowed attention on qkv (f16 WMMA,
//             K/V staged via GLOBAL_LOAD_ASYNC_TO_LDS_B128)       -> ws
//   kernel 3: out = attn @ w_out       (f16 A, f32 B, f32 out -> d_out)
// Workspace layout: [0, 48MB) qkv f16 [8192,3072]; [48MB, 64MB) attn f16 [8192,1024]
// ---------------------------------------------------------------------------

typedef __attribute__((ext_vector_type(16))) _Float16 v16h;
typedef __attribute__((ext_vector_type(8)))  float    v8f;
typedef __attribute__((ext_vector_type(4)))  _Float16 h4;
typedef __attribute__((ext_vector_type(2)))  _Float16 h2;

// A-matrix (16-bit, 16x32) element map: VGPR v (0..7), lane-half hi (0/1)
// -> starting K of the contiguous (K,K+1) pair held in that VGPR.
// Per ISA: VGPR0..3 lanes0-15 K={0,1..6,7}, lanes16-31 K={8..15}; VGPR4..7 +16.
__device__ __forceinline__ int kmapA(int v, int hi) {
  return ((v >> 2) << 4) + ((v & 3) << 1) + (hi << 3);
}
// B-matrix (16-bit, 32x16): same K map as A: VGPR0..3 lanes0-15 K=0..7, lanes16-31 K=8..15; VGPR4..7 +16.
__device__ __forceinline__ int kmapB(int v, int hi) {
  return ((v >> 2) << 4) + ((v & 3) << 1) + (hi << 3);
}

// LDS byte offset of a __shared__-derived pointer (generic -> AS3 -> int).
__device__ __forceinline__ unsigned lds_off(const void* p) {
  return (unsigned)(unsigned long long)(__attribute__((address_space(3))) const void*)p;
}

// Load 16 source elements and convert to f16.
__device__ __forceinline__ void cvt16(const float* p, _Float16* d) {
  const float4* p4 = (const float4*)p;
#pragma unroll
  for (int i = 0; i < 4; i++) {
    float4 t = p4[i];
    d[4 * i + 0] = (_Float16)t.x; d[4 * i + 1] = (_Float16)t.y;
    d[4 * i + 2] = (_Float16)t.z; d[4 * i + 3] = (_Float16)t.w;
  }
}
__device__ __forceinline__ void cvt16(const _Float16* p, _Float16* d) {
  const uint4* p4 = (const uint4*)p;
#pragma unroll
  for (int i = 0; i < 2; i++) {
    union { uint4 u; _Float16 h[8]; } u;
    u.u = p4[i];
#pragma unroll
    for (int j = 0; j < 8; j++) d[8 * i + j] = u.h[j];
  }
}

// ---------------------------------------------------------------------------
// Generic f16-WMMA GEMM: C[M,N] = A[M,K] * B[K,N].  M,N,K multiples of 128/128/32.
// Block tile 128x128x32, 256 threads = 8 waves in a 2(M) x 4(N) grid,
// each wave owns a 64x32 tile = 4x2 WMMA accumulators.
// ---------------------------------------------------------------------------
template <typename TA, typename TO>
__global__ __launch_bounds__(256) void wmma_gemm_kernel(
    const TA* __restrict__ A, const float* __restrict__ B, TO* __restrict__ C,
    int M, int N, int K) {
  constexpr int BM = 128, BN = 128, BK = 32, LDA = 40, LDB = 40;  // +8 pad vs banks
  __shared__ _Float16 Ash[BM * LDA];
  __shared__ _Float16 Bsh[BN * LDB];

  const int tid = threadIdx.x, lane = tid & 31, wave = tid >> 5;
  const int wm = wave >> 2, wn = wave & 3;
  const int hi = lane >> 4, lm = lane & 15;
  const int bm0 = blockIdx.y * BM, bn0 = blockIdx.x * BN;

  v8f acc[4][2] = {};

  const int arow = tid >> 1, acol = (tid & 1) * 16;   // A: 128 rows x 32 k
  const int brow = tid >> 1, bkh  = (tid & 1) * 16;   // B: transpose to [n][k]

  for (int k0 = 0; k0 < K; k0 += BK) {
    __syncthreads();
    {  // stage A tile, row-major [m][k] in f16 (needs f32->f16 convert: VGPR path)
      const TA* ap = A + (size_t)(bm0 + arow) * K + k0 + acol;
      __builtin_prefetch(ap + BK, 0, 0);  // global_prefetch next k-tile
      _Float16 tmp[16] __attribute__((aligned(16)));
      cvt16(ap, tmp);
#pragma unroll
      for (int i = 0; i < 4; i++)
        *(h4*)&Ash[arow * LDA + acol + 4 * i] = *(const h4*)&tmp[4 * i];
    }
    {  // stage B tile transposed: Bsh[n][k]
      const float* bp = B + (size_t)(k0 + bkh) * N + bn0 + brow;
#pragma unroll
      for (int g = 0; g < 4; g++) {
        h4 t;
#pragma unroll
        for (int j = 0; j < 4; j++) t[j] = (_Float16)bp[(size_t)(g * 4 + j) * N];
        *(h4*)&Bsh[brow * LDB + bkh + g * 4] = t;
      }
    }
    __syncthreads();

    v16h af[4], bf[2];
#pragma unroll
    for (int ms = 0; ms < 4; ms++) {
      const _Float16* p = &Ash[(wm * 64 + ms * 16 + lm) * LDA];
#pragma unroll
      for (int v = 0; v < 8; v++) {
        h2 e = *(const h2*)&p[kmapA(v, hi)];
        af[ms][2 * v] = e[0]; af[ms][2 * v + 1] = e[1];
      }
    }
#pragma unroll
    for (int ns = 0; ns < 2; ns++) {
      const _Float16* p = &Bsh[(wn * 32 + ns * 16 + lm) * LDB];
#pragma unroll
      for (int v = 0; v < 8; v++) {
        h2 e = *(const h2*)&p[kmapB(v, hi)];
        bf[ns][2 * v] = e[0]; bf[ns][2 * v + 1] = e[1];
      }
    }
#pragma unroll
    for (int ms = 0; ms < 4; ms++)
#pragma unroll
      for (int ns = 0; ns < 2; ns++)
        acc[ms][ns] = __builtin_amdgcn_wmma_f32_16x16x32_f16(
            false, af[ms], false, bf[ns], (short)0, acc[ms][ns], false, false);
  }

#pragma unroll
  for (int ms = 0; ms < 4; ms++)
#pragma unroll
    for (int ns = 0; ns < 2; ns++)
#pragma unroll
      for (int v = 0; v < 8; v++) {
        int m = bm0 + wm * 64 + ms * 16 + v + 8 * hi;  // C/D layout: M = vgpr + 8*hi
        int n = bn0 + wn * 32 + ns * 16 + lm;          //             N = lane%16
        C[(size_t)m * N + n] = (TO)acc[ms][ns][v];
      }
}

// ---------------------------------------------------------------------------
// Windowed flash attention. One wave per 16-query tile; 4 waves per block.
// qkv f16 [B*T, 3072] (q|k|v), out f16 [B*T, 1024] in [B,T,H*Dh] layout.
// K/V tiles staged with gfx1250 async copy (global -> LDS, no VGPR bounce).
// ---------------------------------------------------------------------------
__global__ __launch_bounds__(128) void sw_attn_kernel(
    const _Float16* __restrict__ qkv, _Float16* __restrict__ out) {
  constexpr int T = 2048, Dh = 64, C = 1024, C3 = 3072, W = 256;
  constexpr int LDK = 72, LDP = 40;
  __shared__ _Float16 KshA[4 * 32 * LDK];
  __shared__ _Float16 VshA[4 * 32 * LDK];
  __shared__ _Float16 PshA[4 * 16 * LDP];

  const int tid = threadIdx.x, lane = tid & 31, wave = tid >> 5;
  const int hi = lane >> 4, lm = lane & 15;
  _Float16* Ksh = KshA + wave * 32 * LDK;
  _Float16* Vsh = VshA + wave * 32 * LDK;
  _Float16* Psh = PshA + wave * 16 * LDP;

  const int b = blockIdx.z, h = blockIdx.y;
  const int q0 = blockIdx.x * 64 + wave * 16;

  // Q as two A-fragments (k = 0..31 and 32..63), gathered straight from global.
  v16h qa[2];
  {
    const _Float16* qb = qkv + (size_t)(b * T + q0 + lm) * C3 + h * Dh;
#pragma unroll
    for (int f = 0; f < 2; f++)
#pragma unroll
      for (int v = 0; v < 8; v++) {
        h2 e = *(const h2*)&qb[f * 32 + kmapA(v, hi)];
        qa[f][2 * v] = e[0]; qa[f][2 * v + 1] = e[1];
      }
  }

  v8f oacc[4] = {};
  float mi[8], li[8];
#pragma unroll
  for (int v = 0; v < 8; v++) { mi[v] = -1e30f; li[v] = 0.0f; }

  int lo = q0 - W; if (lo < 0) lo = 0;
  const int kstart = lo & ~31;
  const int kend = q0 + 15;

  // Per-lane LDS destinations for the async copies (byte offsets in LDS).
  const unsigned kdst = lds_off(&Ksh[lane * LDK]);
  const unsigned vdst = lds_off(&Vsh[lane * LDK]);

  for (int kb = kstart; kb <= kend; kb += 32) {
    {  // async-stage 32 K rows + 32 V rows (f16, 128B each): lane -> one row.
      int j = kb + lane; j = j < (T - 1) ? j : (T - 1);  // clamp; masked below
      const _Float16* kp = qkv + (size_t)(b * T + j) * C3 + C + h * Dh;
      const _Float16* vp = kp + C;
#pragma unroll
      for (int c = 0; c < 8; c++) {
        asm volatile("global_load_async_to_lds_b128 %0, %1, off"
                     :: "v"(kdst + 16u * c),
                        "v"((unsigned long long)(size_t)(kp + 8 * c))
                     : "memory");
        asm volatile("global_load_async_to_lds_b128 %0, %1, off"
                     :: "v"(vdst + 16u * c),
                        "v"((unsigned long long)(size_t)(vp + 8 * c))
                     : "memory");
      }
    }
    asm volatile("s_wait_asynccnt 0x0" ::: "memory");  // LDS data landed
    asm volatile("s_wait_dscnt 0" ::: "memory");

    // S = Q * K^T for two 16-key subtiles (K-dim 64 = two wmma k-steps)
    float sv[2][8];
#pragma unroll
    for (int nt = 0; nt < 2; nt++) {
      v16h kf0, kf1;
      const _Float16* p = &Ksh[(nt * 16 + lm) * LDK];
#pragma unroll
      for (int v = 0; v < 8; v++) {
        int d0 = kmapB(v, hi);
        h2 e0 = *(const h2*)&p[d0];
        h2 e1 = *(const h2*)&p[32 + d0];
        kf0[2 * v] = e0[0]; kf0[2 * v + 1] = e0[1];
        kf1[2 * v] = e1[0]; kf1[2 * v + 1] = e1[1];
      }
      v8f s = {};
      s = __builtin_amdgcn_wmma_f32_16x16x32_f16(false, qa[0], false, kf0, (short)0, s, false, false);
      s = __builtin_amdgcn_wmma_f32_16x16x32_f16(false, qa[1], false, kf1, (short)0, s, false, false);
#pragma unroll
      for (int v = 0; v < 8; v++) {
        int qi = q0 + v + 8 * hi;          // this lane's query row
        int kj = kb + nt * 16 + lm;        // this lane's key column
        float val = s[v] * 0.125f;         // 1/sqrt(64)
        sv[nt][v] = (kj > qi || kj < qi - W) ? -1e30f : val;
      }
    }

    // Online softmax: row stats via 16-lane xor-shuffle reductions (wave32).
#pragma unroll
    for (int v = 0; v < 8; v++) {
      float rm = fmaxf(sv[0][v], sv[1][v]);
#pragma unroll
      for (int d = 8; d >= 1; d >>= 1) rm = fmaxf(rm, __shfl_xor(rm, d, 32));
      float nm = fmaxf(mi[v], rm);
      float p0 = __expf(sv[0][v] - nm);
      float p1 = __expf(sv[1][v] - nm);
      float rs = p0 + p1;
#pragma unroll
      for (int d = 8; d >= 1; d >>= 1) rs += __shfl_xor(rs, d, 32);
      float corr = __expf(mi[v] - nm);
      li[v] = li[v] * corr + rs;
      mi[v] = nm;
#pragma unroll
      for (int c = 0; c < 4; c++) oacc[c][v] *= corr;
      sv[0][v] = p0; sv[1][v] = p1;
    }

    // P (f32 D-layout) -> f16 A-fragment via LDS round trip (same wave, in-order DS).
#pragma unroll
    for (int v = 0; v < 8; v++) {
      int m = v + 8 * hi;
      Psh[m * LDP + lm]      = (_Float16)sv[0][v];
      Psh[m * LDP + 16 + lm] = (_Float16)sv[1][v];
    }
    asm volatile("s_wait_dscnt 0" ::: "memory");
    v16h pa;
    {
      const _Float16* p = &Psh[lm * LDP];
#pragma unroll
      for (int v = 0; v < 8; v++) {
        h2 e = *(const h2*)&p[kmapA(v, hi)];
        pa[2 * v] = e[0]; pa[2 * v + 1] = e[1];
      }
    }

    // O += P * V, one wmma per 16-wide head-dim chunk.
#pragma unroll
    for (int c = 0; c < 4; c++) {
      v16h vf;
#pragma unroll
      for (int v = 0; v < 8; v++) {
        int kk = kmapB(v, hi);
        vf[2 * v]     = Vsh[(size_t)kk * LDK + c * 16 + lm];
        vf[2 * v + 1] = Vsh[(size_t)(kk + 1) * LDK + c * 16 + lm];
      }
      oacc[c] = __builtin_amdgcn_wmma_f32_16x16x32_f16(
          false, pa, false, vf, (short)0, oacc[c], false, false);
    }
  }

  // Normalize and store in [B,T,H*Dh] layout for the output projection.
#pragma unroll
  for (int c = 0; c < 4; c++)
#pragma unroll
    for (int v = 0; v < 8; v++) {
      int m = v + 8 * hi;
      int t = q0 + m;
      out[(size_t)(b * T + t) * C + h * Dh + c * 16 + lm] =
          (_Float16)(oacc[c][v] / li[v]);
    }
}

// ---------------------------------------------------------------------------
extern "C" void kernel_launch(void* const* d_in, const int* in_sizes, int n_in,
                              void* d_out, int out_size, void* d_ws, size_t ws_size,
                              hipStream_t stream) {
  (void)in_sizes; (void)n_in; (void)out_size; (void)ws_size;
  const float* x      = (const float*)d_in[0];  // [4,2048,1024]
  const float* w_qkv  = (const float*)d_in[1];  // [1024,3072]
  const float* w_out  = (const float*)d_in[2];  // [1024,1024]
  float* outp         = (float*)d_out;          // [4,2048,1024]

  constexpr int M = 4 * 2048, K = 1024, N3 = 3072, N1 = 1024;
  _Float16* qkv  = (_Float16*)d_ws;                                    // 48 MB
  _Float16* attn = (_Float16*)((char*)d_ws + (size_t)M * N3 * 2);      // 16 MB

  dim3 g1(N3 / 128, M / 128);
  wmma_gemm_kernel<float, _Float16><<<g1, 256, 0, stream>>>(x, w_qkv, qkv, M, N3, K);

  dim3 ga(2048 / 64, 16, 4);  // (query blocks, heads, batch)
  sw_attn_kernel<<<ga, 128, 0, stream>>>(qkv, attn);

  dim3 g2(N1 / 128, M / 128);
  wmma_gemm_kernel<_Float16, float><<<g2, 256, 0, stream>>>(attn, w_out, outp, M, N1, K);
}
